// Block_29583734734992
// MI455X (gfx1250) — matrix operand order repaired and checked by reading, not compiled, not run on hardware
//
#include <hip/hip_runtime.h>

typedef __attribute__((ext_vector_type(16))) __bf16 v16bf;
typedef __attribute__((ext_vector_type(8)))  __bf16 v8bf;
typedef __attribute__((ext_vector_type(8)))  float  v8f;
typedef __attribute__((ext_vector_type(4)))  unsigned int v4u;
typedef __attribute__((ext_vector_type(8)))  int v8i;
typedef __attribute__((ext_vector_type(4)))  int v4i;

#define D_MODEL 1024
#define N_HEADS 16
#define D_HEAD  64
#define D_MLP   4096
#define BATCH   2
#define SEQ     2048
#define NTOK    (BATCH*SEQ)

static __device__ __forceinline__ v16bf cat8(v8bf lo, v8bf hi) {
  return __builtin_shufflevector(lo, hi, 0,1,2,3,4,5,6,7,8,9,10,11,12,13,14,15);
}

static __device__ __forceinline__ float gelu_f(float x) {
  const float c = 0.7978845608028654f;
  return 0.5f * x * (1.f + tanhf(c * (x + 0.044715f * x * x * x)));
}

// ---------------------------------------------------------------------------
// TDM: async-load a 2D bf16 tile [rows x row_elems] (row stride = stride_elems)
// from global into LDS at lds_addr, packed contiguously (row_elems*2 B/row).
// D# layout per CDNA5 ISA 8.3/8.4: group0 {count=1, lds_addr, global_addr,
// type=2}; group1 {data_size=1(2B), tensor dims (huge, no OOB clamp), tile
// dims, dim0 stride}; groups 2/3 zero (2D tile). All operands block-uniform
// -> SGPRs, as the VIMAGE tensor encoding requires. This toolchain exposes
// the 6-arg builtin (extra int32x8 group, unused here -> zeros).
// ---------------------------------------------------------------------------
static __device__ __forceinline__ void tdm_load_2d(
    const void* gaddr, unsigned lds_addr, unsigned rows, unsigned row_elems,
    unsigned long long stride_elems) {
  const unsigned long long ga = (unsigned long long)gaddr;
  const unsigned td0 = 0x100000u, td1 = 0x100000u;  // huge tensor dims
  v4u g0;
  g0[0] = 1u;                                       // count=1, user mode
  g0[1] = lds_addr;                                 // LDS byte address
  g0[2] = (unsigned)(ga & 0xffffffffu);             // global_addr[31:0]
  g0[3] = (unsigned)((ga >> 32) & 0x01ffffffu) | (2u << 30);  // [56:32]|type=2
  v8i g1;
  g1[0] = (int)(1u << 16);                          // data_size=1 (2 bytes)
  g1[1] = (int)(td0 << 16);                         // tensor_dim0[15:0]
  g1[2] = (int)((td0 >> 16) | (td1 << 16));         // dim0[31:16] | dim1[15:0]
  g1[3] = (int)((td1 >> 16) | (row_elems << 16));   // dim1[31:16] | tile_dim0
  g1[4] = (int)rows;                                // tile_dim1 (tile_dim2=0)
  g1[5] = (int)(stride_elems & 0xffffffffu);        // dim0_stride[31:0]
  g1[6] = (int)((stride_elems >> 32) & 0xffffu);    // dim0_stride[47:32]
  g1[7] = 0;
  v4i g2 = {0, 0, 0, 0};
  v4i g3 = {0, 0, 0, 0};
  v8i g4 = {0, 0, 0, 0, 0, 0, 0, 0};
  __builtin_amdgcn_tensor_load_to_lds(g0, g1, g2, g3, g4, 0);
}

// ---------------------------------------------------------------------------
// fp32 [K][N] -> bf16 [N][K]  (batched over blockIdx.z, stride K*N)
// ---------------------------------------------------------------------------
__global__ __launch_bounds__(256) void transpose_cvt_kernel(
    const float* __restrict__ src, __bf16* __restrict__ dst, int K, int N) {
  long boff = (long)blockIdx.z * K * N;
  int idx = blockIdx.x * 256 + threadIdx.x;
  if (idx < K * N) {
    int k = idx / N, n = idx % N;
    dst[boff + (long)n * K + k] = (__bf16)src[boff + idx];
  }
}

// ---------------------------------------------------------------------------
// LayerNorm: fp32 [rows][1024] -> bf16 [rows][1024]
// ---------------------------------------------------------------------------
__global__ __launch_bounds__(256) void ln_kernel(
    const float* __restrict__ x, const float* __restrict__ w,
    const float* __restrict__ b, __bf16* __restrict__ out) {
  __shared__ float s1[256], s2[256];
  const int row = blockIdx.x, tid = threadIdx.x;
  const float* xr = x + (long)row * D_MODEL;
  float v[4], sum = 0.f, sq = 0.f;
#pragma unroll
  for (int i = 0; i < 4; i++) { v[i] = xr[tid + i * 256]; sum += v[i]; sq += v[i] * v[i]; }
  s1[tid] = sum; s2[tid] = sq; __syncthreads();
  for (int s = 128; s > 0; s >>= 1) {
    if (tid < s) { s1[tid] += s1[tid + s]; s2[tid] += s2[tid + s]; }
    __syncthreads();
  }
  const float mean = s1[0] * (1.f / D_MODEL);
  const float var  = s2[0] * (1.f / D_MODEL) - mean * mean;
  const float rstd = rsqrtf(var + 1e-5f);
#pragma unroll
  for (int i = 0; i < 4; i++) {
    int c = tid + i * 256;
    out[(long)row * D_MODEL + c] = (__bf16)((v[i] - mean) * rstd * w[c] + b[c]);
  }
}

// ---------------------------------------------------------------------------
// bf16 WMMA GEMM: C[M,N] = A[M,K] * Bt[N,K]^T  (Bt pre-transposed)
// Block tile 128x64, 8 waves, wave tile 32x32 (2x2 fragments), BK=32.
// Tiles staged LDS via Tensor Data Mover (wave 0 issues, s_wait_tensorcnt).
// EPI: 0 = bf16 out (scaled, alpha)      [Q/K projection]
//      1 = bf16 out transposed           [V -> Vt]
//      2 = f32  out = fadd + C           [attn out-proj + residual]
//      3 = bf16 out = gelu(C + bias)     [MLP in]
//      4 = f32  out = fadd + C + bias    [MLP out + residual]
// ---------------------------------------------------------------------------
template <int EPI>
__global__ __launch_bounds__(256) void gemm_bf16_wmma(
    const __bf16* __restrict__ A, int lda,
    const __bf16* __restrict__ Bt, int ldb, int K,
    float* __restrict__ fout, const float* __restrict__ fadd,
    const float* __restrict__ bias, __bf16* __restrict__ bout,
    int ldo, float alpha,
    int aDiv, long aBatch, int bMod, long bBatch, long oBatch) {
  __shared__ __attribute__((aligned(32))) __bf16 As[128 * 32];
  __shared__ __attribute__((aligned(32))) __bf16 Bs[64 * 32];
  const int tid  = threadIdx.x;
  const int wave = tid >> 5, lane = tid & 31;
  const int wm = wave >> 1, wn = wave & 1;
  const int row = lane & 15, hi = lane >> 4, kb = hi * 8;
  const int m0 = blockIdx.x * 128, n0 = blockIdx.y * 64;

  A  += (long)(blockIdx.z / aDiv) * aBatch;
  Bt += (long)(blockIdx.z % bMod) * bBatch;
  const long oOff = (long)blockIdx.z * oBatch;

  const unsigned ldsA = (unsigned)(uintptr_t)(void*)As;
  const unsigned ldsB = (unsigned)(uintptr_t)(void*)Bs;

  v8f acc[2][2] = {};
  for (int k0 = 0; k0 < K; k0 += 32) {
    if (wave == 0) {
      tdm_load_2d(&A[(long)m0 * lda + k0], ldsA, 128, 32, (unsigned)lda);
      tdm_load_2d(&Bt[(long)n0 * ldb + k0], ldsB, 64, 32, (unsigned)ldb);
      __builtin_amdgcn_s_wait_tensorcnt(0);
    }
    __syncthreads();

    v16bf af[2], bfr[2];
#pragma unroll
    for (int i = 0; i < 2; i++) {
      const __bf16* p = &As[(wm * 32 + i * 16 + row) * 32];
      af[i] = *(const v16bf*)&p[hi * 16];
    }
#pragma unroll
    for (int j = 0; j < 2; j++)
      bfr[j] = *(const v16bf*)&Bs[(wn * 32 + j * 16 + row) * 32 + hi * 16];
#pragma unroll
    for (int i = 0; i < 2; i++)
#pragma unroll
      for (int j = 0; j < 2; j++)
        acc[i][j] = __builtin_amdgcn_wmma_f32_16x16x32_bf16(
            false, af[i], false, bfr[j], (short)0, acc[i][j], false, false);
    __syncthreads();
  }

#pragma unroll
  for (int i = 0; i < 2; i++)
#pragma unroll
    for (int j = 0; j < 2; j++)
#pragma unroll
      for (int r = 0; r < 8; r++) {
        const int mi = m0 + wm * 32 + i * 16 + r + hi * 8;
        const int ni = n0 + wn * 32 + j * 16 + row;
        const float c = acc[i][j][r];
        if (EPI == 0) {
          bout[oOff + (long)mi * ldo + ni] = (__bf16)(alpha * c);
        } else if (EPI == 1) {
          bout[oOff + (long)ni * ldo + mi] = (__bf16)c;
        } else if (EPI == 2) {
          const long idx = (long)mi * ldo + ni;
          fout[idx] = fadd[idx] + c;
        } else if (EPI == 3) {
          bout[(long)mi * ldo + ni] = (__bf16)gelu_f(c + bias[ni]);
        } else {
          const long idx = (long)mi * ldo + ni;
          fout[idx] = fadd[idx] + c + bias[ni];
        }
      }
}

// ---------------------------------------------------------------------------
// Flash attention, causal. Grid: (SEQ/64, BATCH*N_HEADS), 128 threads.
// Each wave owns 16 query rows; 32 keys per iteration:
//   4 score WMMAs (QK^T, dh=64) + online softmax + 4 PV WMMAs.
// Q pre-scaled by 1/sqrt(dh); Vt is [bh][e][s].
// ---------------------------------------------------------------------------
__global__ __launch_bounds__(128) void attn_kernel(
    const __bf16* __restrict__ q, const __bf16* __restrict__ k,
    const __bf16* __restrict__ vt, __bf16* __restrict__ z) {
  __shared__ __attribute__((aligned(32))) __bf16 Pst[4][16 * 32];
  const int bh = blockIdx.y;
  const int b = bh >> 4, h = bh & 15;
  const __bf16* Q  = q  + (long)bh * SEQ * D_HEAD;
  const __bf16* Kb = k  + (long)bh * SEQ * D_HEAD;
  const __bf16* Vt = vt + (long)bh * D_HEAD * SEQ;
  const int wave = threadIdx.x >> 5, lane = threadIdx.x & 31;
  const int row = lane & 15, hi = lane >> 4, kbo = hi * 8;
  const int q0 = blockIdx.x * 64 + wave * 16;

  v16bf qa[2];
#pragma unroll
  for (int t = 0; t < 2; t++) {
    const __bf16* p = &Q[(long)(q0 + row) * D_HEAD + t * 32];
    qa[t] = *(const v16bf*)&p[hi * 16];
  }
  float mrow[8], lrow[8];
#pragma unroll
  for (int r = 0; r < 8; r++) { mrow[r] = -1e30f; lrow[r] = 0.f; }
  v8f o[4] = {};
  __bf16* Pw = &Pst[wave][0];

  const int nkt = (q0 + 16 + 31) >> 5;
  for (int kt32 = 0; kt32 < nkt; ++kt32) {
    const int kt = kt32 * 32;
    v8f s[2] = {};
#pragma unroll
    for (int j = 0; j < 2; j++) {
      const __bf16* kp = &Kb[(long)(kt + j * 16 + row) * D_HEAD];
      v16bf b0 = *(const v16bf*)&kp[hi * 16];
      v16bf b1 = *(const v16bf*)&kp[32 + hi * 16];
      s[j] = __builtin_amdgcn_wmma_f32_16x16x32_bf16(false, qa[0], false, b0,
                                                     (short)0, s[j], false, false);
      s[j] = __builtin_amdgcn_wmma_f32_16x16x32_bf16(false, qa[1], false, b1,
                                                     (short)0, s[j], false, false);
    }
#pragma unroll
    for (int r = 0; r < 8; r++) {
      const int qpos = q0 + r + hi * 8;
      float v0 = (kt + row      > qpos) ? -1e30f : s[0][r];
      float v1 = (kt + 16 + row > qpos) ? -1e30f : s[1][r];
      float mx = fmaxf(v0, v1);
#pragma unroll
      for (int off = 8; off >= 1; off >>= 1) mx = fmaxf(mx, __shfl_xor(mx, off, 16));
      const float mnew = fmaxf(mrow[r], mx);
      const float a = __expf(mrow[r] - mnew);
      mrow[r] = mnew;
      const float p0 = __expf(v0 - mnew);
      const float p1 = __expf(v1 - mnew);
      float rs = p0 + p1;
#pragma unroll
      for (int off = 8; off >= 1; off >>= 1) rs += __shfl_xor(rs, off, 16);
      lrow[r] = lrow[r] * a + rs;
#pragma unroll
      for (int e = 0; e < 4; e++) o[e][r] *= a;
      Pw[(r + hi * 8) * 32 + row]      = (__bf16)p0;
      Pw[(r + hi * 8) * 32 + 16 + row] = (__bf16)p1;
    }
    // C-fragment -> A-fragment relayout through wave-private LDS
    v16bf pa = *(const v16bf*)&Pw[row * 32 + hi * 16];
#pragma unroll
    for (int e = 0; e < 4; e++) {
      v16bf vb = *(const v16bf*)&Vt[(long)(e * 16 + row) * SEQ + kt + hi * 16];
      o[e] = __builtin_amdgcn_wmma_f32_16x16x32_bf16(false, pa, false, vb,
                                                     (short)0, o[e], false, false);
    }
  }
#pragma unroll
  for (int r = 0; r < 8; r++) {
    const float linv = 1.f / lrow[r];
    const long zr = (long)(b * SEQ + q0 + r + hi * 8) * D_MODEL + h * D_HEAD;
#pragma unroll
    for (int e = 0; e < 4; e++)
      z[zr + e * 16 + row] = (__bf16)(o[e][r] * linv);
  }
}

// ---------------------------------------------------------------------------
extern "C" void kernel_launch(void* const* d_in, const int* in_sizes, int n_in,
                              void* d_out, int out_size, void* d_ws, size_t ws_size,
                              hipStream_t stream) {
  (void)in_sizes; (void)n_in; (void)out_size; (void)ws_size;
  const float* resid = (const float*)d_in[0];
  const float* w_q   = (const float*)d_in[1];
  const float* w_k   = (const float*)d_in[2];
  const float* w_v   = (const float*)d_in[3];
  const float* w_o   = (const float*)d_in[4];
  const float* ln1_w = (const float*)d_in[5];
  const float* ln1_b = (const float*)d_in[6];
  const float* ln2_w = (const float*)d_in[7];
  const float* ln2_b = (const float*)d_in[8];
  const float* w_in  = (const float*)d_in[9];
  const float* b_in  = (const float*)d_in[10];
  const float* w_out = (const float*)d_in[11];
  const float* b_out = (const float*)d_in[12];
  float* out = (float*)d_out;

  char* ws = (char*)d_ws;
  auto alloc = [&](size_t bytes) {
    char* p = ws; ws += (bytes + 255) & ~(size_t)255; return p;
  };
  __bf16* wqT   = (__bf16*)alloc((size_t)N_HEADS * D_HEAD * D_MODEL * 2);
  __bf16* wkT   = (__bf16*)alloc((size_t)N_HEADS * D_HEAD * D_MODEL * 2);
  __bf16* wvT   = (__bf16*)alloc((size_t)N_HEADS * D_HEAD * D_MODEL * 2);
  __bf16* woT   = (__bf16*)alloc((size_t)D_MODEL * D_MODEL * 2);
  __bf16* winT  = (__bf16*)alloc((size_t)D_MLP * D_MODEL * 2);
  __bf16* woutT = (__bf16*)alloc((size_t)D_MODEL * D_MLP * 2);
  __bf16* xln1  = (__bf16*)alloc((size_t)NTOK * D_MODEL * 2);
  __bf16* qb    = (__bf16*)alloc((size_t)BATCH * N_HEADS * SEQ * D_HEAD * 2);
  __bf16* kbuf  = (__bf16*)alloc((size_t)BATCH * N_HEADS * SEQ * D_HEAD * 2);
  __bf16* vtb   = (__bf16*)alloc((size_t)BATCH * N_HEADS * D_HEAD * SEQ * 2);
  __bf16* zb    = (__bf16*)alloc((size_t)NTOK * D_MODEL * 2);
  float*  rmid  = (float*) alloc((size_t)NTOK * D_MODEL * 4);
  __bf16* xln2  = (__bf16*)alloc((size_t)NTOK * D_MODEL * 2);
  __bf16* h1    = (__bf16*)alloc((size_t)NTOK * D_MLP * 2);

  // weight convert + transpose to [N][K] bf16
  transpose_cvt_kernel<<<dim3((D_MODEL*D_HEAD)/256, 1, N_HEADS), 256, 0, stream>>>(w_q, wqT, D_MODEL, D_HEAD);
  transpose_cvt_kernel<<<dim3((D_MODEL*D_HEAD)/256, 1, N_HEADS), 256, 0, stream>>>(w_k, wkT, D_MODEL, D_HEAD);
  transpose_cvt_kernel<<<dim3((D_MODEL*D_HEAD)/256, 1, N_HEADS), 256, 0, stream>>>(w_v, wvT, D_MODEL, D_HEAD);
  transpose_cvt_kernel<<<dim3((D_MODEL*D_MODEL)/256, 1, 1), 256, 0, stream>>>(w_o, woT, D_MODEL, D_MODEL);
  transpose_cvt_kernel<<<dim3((D_MODEL*D_MLP)/256, 1, 1), 256, 0, stream>>>(w_in, winT, D_MODEL, D_MLP);
  transpose_cvt_kernel<<<dim3((D_MLP*D_MODEL)/256, 1, 1), 256, 0, stream>>>(w_out, woutT, D_MLP, D_MODEL);

  // LN1
  ln_kernel<<<NTOK, 256, 0, stream>>>(resid, ln1_w, ln1_b, xln1);

  // QKV projections (batched per b,h on grid.z); Q scaled by 1/sqrt(64)
  gemm_bf16_wmma<0><<<dim3(SEQ/128, 1, BATCH*N_HEADS), 256, 0, stream>>>(
      xln1, D_MODEL, wqT, D_MODEL, D_MODEL, nullptr, nullptr, nullptr, qb,
      D_HEAD, 0.125f, N_HEADS, (long)SEQ*D_MODEL, N_HEADS, (long)D_HEAD*D_MODEL,
      (long)SEQ*D_HEAD);
  gemm_bf16_wmma<0><<<dim3(SEQ/128, 1, BATCH*N_HEADS), 256, 0, stream>>>(
      xln1, D_MODEL, wkT, D_MODEL, D_MODEL, nullptr, nullptr, nullptr, kbuf,
      D_HEAD, 1.0f, N_HEADS, (long)SEQ*D_MODEL, N_HEADS, (long)D_HEAD*D_MODEL,
      (long)SEQ*D_HEAD);
  gemm_bf16_wmma<1><<<dim3(SEQ/128, 1, BATCH*N_HEADS), 256, 0, stream>>>(
      xln1, D_MODEL, wvT, D_MODEL, D_MODEL, nullptr, nullptr, nullptr, vtb,
      SEQ, 1.0f, N_HEADS, (long)SEQ*D_MODEL, N_HEADS, (long)D_HEAD*D_MODEL,
      (long)D_HEAD*SEQ);

  // causal flash attention -> z [b][s][h*64] bf16
  attn_kernel<<<dim3(SEQ/64, BATCH*N_HEADS), 128, 0, stream>>>(qb, kbuf, vtb, zb);

  // out-projection + residual -> rmid (fp32)
  gemm_bf16_wmma<2><<<dim3(NTOK/128, D_MODEL/64, 1), 256, 0, stream>>>(
      zb, D_MODEL, woT, D_MODEL, D_MODEL, rmid, resid, nullptr, nullptr,
      D_MODEL, 1.0f, 1, 0, 1, 0, 0);

  // LN2
  ln_kernel<<<NTOK, 256, 0, stream>>>(rmid, ln2_w, ln2_b, xln2);

  // MLP in: gelu(x*w_in + b_in) -> h1 bf16
  gemm_bf16_wmma<3><<<dim3(NTOK/128, D_MLP/64, 1), 256, 0, stream>>>(
      xln2, D_MODEL, winT, D_MODEL, D_MODEL, nullptr, nullptr, b_in, h1,
      D_MLP, 1.0f, 1, 0, 1, 0, 0);

  // MLP out + residual + b_out -> d_out fp32
  gemm_bf16_wmma<4><<<dim3(NTOK/128, D_MODEL/64, 1), 256, 0, stream>>>(
      h1, D_MLP, woutT, D_MLP, D_MLP, out, rmid, b_out, nullptr,
      D_MODEL, 1.0f, 1, 0, 1, 0, 0);
}
